// GAT_16045997818077
// MI455X (gfx1250) — compile-verified, matrix operand order checked by reading, not run on hardware
//
#include <hip/hip_runtime.h>
#include <hip/hip_bf16.h>

typedef __bf16 bf16_t;
typedef __attribute__((ext_vector_type(16))) __bf16 v16bf;
typedef __attribute__((ext_vector_type(8)))  __bf16 v8bf;
typedef __attribute__((ext_vector_type(8)))  float  v8f;

union ABFrag { v16bf v; v8bf h[2]; };

// ---------------------------------------------------------------------------
// elementwise f32 -> bf16
__global__ void k_f32_to_bf16(const float* __restrict__ in, bf16_t* __restrict__ out, int count) {
    int i = blockIdx.x * blockDim.x + threadIdx.x;
    if (i < count) out[i] = (bf16_t)in[i];
}

// [R,C] f32 -> [C,R] bf16
__global__ void k_transpose_f32_to_bf16(const float* __restrict__ in, bf16_t* __restrict__ out,
                                        int R, int C) {
    int idx = blockIdx.x * blockDim.x + threadIdx.x;
    if (idx >= R * C) return;
    int r = idx / C, c = idx - r * C;
    out[(size_t)c * R + r] = (bf16_t)in[(size_t)r * C + c];
}

// ---------------------------------------------------------------------------
// C[M,N] f32 = A[M,K] bf16 (row major) * Bt[N,K] bf16 (row major, i.e. B^T)
// block = 128 threads = 4 waves; each wave -> one 16x16 C tile; grid = (N/64, M/16)
// A-frag 16-bit layout: lane m (0..15,half=0) holds K {8h..8h+7, 16+8h..16+8h+7}
__global__ void k_gemm_bf16_wmma(const bf16_t* __restrict__ A, const bf16_t* __restrict__ Bt,
                                 float* __restrict__ C, int M, int N, int K) {
    const int lane = threadIdx.x & 31;
    const int wave = threadIdx.x >> 5;
    const int m0 = blockIdx.y * 16;
    const int n0 = blockIdx.x * 64 + wave * 16;
    const int m = lane & 15, half = lane >> 4;

    const bf16_t* arow = A  + (size_t)(m0 + m) * K;
    const bf16_t* brow = Bt + (size_t)(n0 + m) * K;

    v8f acc = {};
    for (int k0 = 0; k0 < K; k0 += 32) {
        ABFrag a, b;
        a.h[0] = *(const v8bf*)(arow + k0 + 8 * half);
        a.h[1] = *(const v8bf*)(arow + k0 + 16 + 8 * half);
        b.h[0] = *(const v8bf*)(brow + k0 + 8 * half);
        b.h[1] = *(const v8bf*)(brow + k0 + 16 + 8 * half);
        acc = __builtin_amdgcn_wmma_f32_16x16x32_bf16(false, a.v, false, b.v,
                                                      (short)0, acc, false, false);
    }
#pragma unroll
    for (int r = 0; r < 8; ++r)
        C[(size_t)(m0 + r + 8 * half) * N + n0 + m] = acc[r];
}

// ---------------------------------------------------------------------------
// el/er scores, stored transposed [H, n] for coalesced j-direction loads later
__global__ void k_scores(const float* __restrict__ g, const float* __restrict__ al,
                         const float* __restrict__ ar, float* __restrict__ elT,
                         float* __restrict__ erT, int n, int H, int F) {
    int idx = blockIdx.x * blockDim.x + threadIdx.x;
    if (idx >= n * H) return;
    int i = idx / H, h = idx - i * H;
    const float* gp = g + (size_t)i * H * F + (size_t)h * F;
    float el = 0.f, er = 0.f;
    for (int f = 0; f < F; ++f) { float v = gp[f]; el += v * al[f]; er += v * ar[f]; }
    elT[(size_t)h * n + i] = el;
    erT[(size_t)h * n + i] = er;
}

// ---------------------------------------------------------------------------
// Fused GAT attention: per 16-row i-tile, stream j in chunks of 32.
// P tile (attention numerators) built in registers in WMMA A-fragment layout,
// multiplied against Gt (bf16 [H*F, n]) B-fragments; row sums via shfl_xor(16).
// out_bf != nullptr : write bf16 with ELU (layer 1).  else f32 (layer 2).
// grid = (n/16, H), block = (F/16)*32 threads.
__global__ void k_gat_aggregate(const int* __restrict__ adj,
                                const float* __restrict__ elT, const float* __restrict__ erT,
                                const bf16_t* __restrict__ Gt,
                                bf16_t* __restrict__ out_bf, float* __restrict__ out_f32,
                                int n, int H, int F) {
    const int lane = threadIdx.x & 31;
    const int wave = threadIdx.x >> 5;
    const int h    = blockIdx.y;
    const int i0   = blockIdx.x * 16;
    const int m = lane & 15, half = lane >> 4;
    const int i    = i0 + m;
    const int Ftot = H * F;
    const int f0   = h * F + wave * 16;

    const float el_i = elT[(size_t)h * n + i];
    const int*    arow = adj + (size_t)i * n;
    const float*  erow = erT + (size_t)h * n;
    const bf16_t* grow = Gt  + (size_t)(f0 + m) * n;

    v8f acc = {};
    float rowsum = 0.f;

    auto sc = [&](int ad, float er) -> float {
        float t = el_i + er;
        t = t > 0.f ? t : 0.2f * t;             // LeakyReLU(0.2)
        float p = ad ? __expf(t) : 0.f;         // mask -> exp(-inf) == 0
        rowsum += p;
        return p;
    };

    for (int jt = 0; jt < n; jt += 32) {
        const int jlo = jt + 8 * half;
        const int jhi = jlo + 16;
        __builtin_prefetch(arow + jlo + 64, 0, 0);   // stream next adj chunk

        int4 a0 = *(const int4*)(arow + jlo);
        int4 a1 = *(const int4*)(arow + jlo + 4);
        int4 a2 = *(const int4*)(arow + jhi);
        int4 a3 = *(const int4*)(arow + jhi + 4);
        float4 e0 = *(const float4*)(erow + jlo);
        float4 e1 = *(const float4*)(erow + jlo + 4);
        float4 e2 = *(const float4*)(erow + jhi);
        float4 e3 = *(const float4*)(erow + jhi + 4);

        float pv[16];
        pv[0]  = sc(a0.x, e0.x); pv[1]  = sc(a0.y, e0.y);
        pv[2]  = sc(a0.z, e0.z); pv[3]  = sc(a0.w, e0.w);
        pv[4]  = sc(a1.x, e1.x); pv[5]  = sc(a1.y, e1.y);
        pv[6]  = sc(a1.z, e1.z); pv[7]  = sc(a1.w, e1.w);
        pv[8]  = sc(a2.x, e2.x); pv[9]  = sc(a2.y, e2.y);
        pv[10] = sc(a2.z, e2.z); pv[11] = sc(a2.w, e2.w);
        pv[12] = sc(a3.x, e3.x); pv[13] = sc(a3.y, e3.y);
        pv[14] = sc(a3.z, e3.z); pv[15] = sc(a3.w, e3.w);

        ABFrag af, bfr;
#pragma unroll
        for (int t = 0; t < 16; ++t) af.v[t] = (bf16_t)pv[t];
        bfr.h[0] = *(const v8bf*)(grow + jlo);
        bfr.h[1] = *(const v8bf*)(grow + jhi);

        acc = __builtin_amdgcn_wmma_f32_16x16x32_bf16(false, af.v, false, bfr.v,
                                                      (short)0, acc, false, false);
    }

    // combine the two half-row partial sums (wave32)
    rowsum += __shfl_xor(rowsum, 16, 32);

#pragma unroll
    for (int r = 0; r < 8; ++r) {
        float s = __shfl(rowsum, r + 8 * half, 32);   // row sum for C-row r+8*half
        float o = acc[r] / s;
        int row = i0 + r + 8 * half;
        int col = f0 + m;
        if (out_bf) {
            float e = o > 0.f ? o : (__expf(o) - 1.f);   // ELU
            out_bf[(size_t)row * Ftot + col] = (bf16_t)e;
        } else {
            out_f32[(size_t)row * Ftot + col] = o;
        }
    }
}

// ---------------------------------------------------------------------------
extern "C" void kernel_launch(void* const* d_in, const int* in_sizes, int n_in,
                              void* d_out, int out_size, void* d_ws, size_t ws_size,
                              hipStream_t stream) {
    const int N = 2048, IN = 512, HID = 256, OUT = 128, H = 8;
    const int F1 = HID / H;   // 32

    const float* x    = (const float*)d_in[0];
    const float* W1   = (const float*)d_in[1];
    const float* a1_l = (const float*)d_in[2];
    const float* a1_r = (const float*)d_in[3];
    const float* W2   = (const float*)d_in[4];
    const float* a2_l = (const float*)d_in[5];
    const float* a2_r = (const float*)d_in[6];
    const int*   adj  = (const int*)d_in[7];
    float* out = (float*)d_out;

    char* ws = (char*)d_ws;
    size_t off = 0;
    auto alloc = [&](size_t bytes) -> char* {
        char* p = ws + off;
        off = (off + bytes + 255) & ~(size_t)255;
        return p;
    };
    bf16_t* xb   = (bf16_t*)alloc((size_t)N * IN * 2);
    bf16_t* w1t  = (bf16_t*)alloc((size_t)HID * IN * 2);
    bf16_t* w2t  = (bf16_t*)alloc((size_t)OUT * HID * 2);
    float*  g1   = (float*) alloc((size_t)N * HID * 4);
    bf16_t* g1t  = (bf16_t*)alloc((size_t)HID * N * 2);
    float*  elT1 = (float*) alloc((size_t)H * N * 4);
    float*  erT1 = (float*) alloc((size_t)H * N * 4);
    bf16_t* h1b  = (bf16_t*)alloc((size_t)N * HID * 2);
    float*  g2   = (float*) alloc((size_t)N * OUT * 4);
    bf16_t* g2t  = (bf16_t*)alloc((size_t)OUT * N * 2);
    float*  elT2 = (float*) alloc((size_t)N * 4);
    float*  erT2 = (float*) alloc((size_t)N * 4);

    // --- operand prep (bf16 + transposes) ---
    k_f32_to_bf16<<<dim3((N * IN + 255) / 256), 256, 0, stream>>>(x, xb, N * IN);
    k_transpose_f32_to_bf16<<<dim3((IN * HID + 255) / 256), 256, 0, stream>>>(W1, w1t, IN, HID);
    k_transpose_f32_to_bf16<<<dim3((HID * OUT + 255) / 256), 256, 0, stream>>>(W2, w2t, HID, OUT);

    // --- layer 1 ---
    k_gemm_bf16_wmma<<<dim3(HID / 64, N / 16), 128, 0, stream>>>(xb, w1t, g1, N, HID, IN);
    k_scores<<<dim3((N * H + 255) / 256), 256, 0, stream>>>(g1, a1_l, a1_r, elT1, erT1, N, H, F1);
    k_transpose_f32_to_bf16<<<dim3((N * HID + 255) / 256), 256, 0, stream>>>(g1, g1t, N, HID);
    k_gat_aggregate<<<dim3(N / 16, H), (F1 / 16) * 32, 0, stream>>>(
        adj, elT1, erT1, g1t, h1b, nullptr, N, H, F1);

    // --- layer 2 ---
    k_gemm_bf16_wmma<<<dim3(OUT / 64, N / 16), 128, 0, stream>>>(h1b, w2t, g2, N, OUT, HID);
    k_scores<<<dim3((N + 255) / 256), 256, 0, stream>>>(g2, a2_l, a2_r, elT2, erT2, N, 1, OUT);
    k_transpose_f32_to_bf16<<<dim3((N * OUT + 255) / 256), 256, 0, stream>>>(g2, g2t, N, OUT);
    k_gat_aggregate<<<dim3(N / 16, 1), (OUT / 16) * 32, 0, stream>>>(
        adj, elT2, erT2, g2t, nullptr, out, N, 1, OUT);
}
